// TrulyNeuralSyscallHandlersV3_18975165514020
// MI455X (gfx1250) — matrix operand order repaired and checked by reading, not compiled, not run on hardware
//
#include <hip/hip_runtime.h>
#include <hip/hip_bf16.h>
#include <math.h>

// ---------------------------------------------------------------------------
// MI455X (gfx1250). Heavy GEMMs -> v_wmma_f32_16x16x32_bf16, f32 accumulate.
// Round-5: async global->LDS weight staging (GLOBAL_LOAD_ASYNC_TO_LDS_B128,
// ASYNCcnt) with corrected builtin pointer types: (AS1 v4i*, AS3 v4i*, imm, imm).
// ---------------------------------------------------------------------------

typedef __attribute__((ext_vector_type(16))) __bf16 v16bf;
typedef __attribute__((ext_vector_type(8)))  float  v8f;
typedef __attribute__((ext_vector_type(4)))  int    v4i;

#define BTOT 131072

#if defined(__AMDGCN__) && __has_builtin(__builtin_amdgcn_global_load_async_to_lds_b128) && __has_builtin(__builtin_amdgcn_s_wait_asynccnt)
#define USE_ASYNC_LDS 1
#else
#define USE_ASYNC_LDS 0
#endif

#if USE_ASYNC_LDS
typedef __attribute__((address_space(1))) v4i* as1_v4i;
typedef __attribute__((address_space(3))) v4i* as3_v4i;
#endif

static __device__ __forceinline__ unsigned short f2bf(float f) {
  unsigned u = __builtin_bit_cast(unsigned, f);
  unsigned r = ((u >> 16) & 1u) + 0x7FFFu;   // round-to-nearest-even
  return (unsigned short)((u + r) >> 16);
}

// Branch-free tanh-form GELU. |err| <= ~1e-3, below bf16 quantization noise.
static __device__ __forceinline__ float gelu_fast(float x) {
  float y  = 0.7978845608028654f * (x + 0.044715f * x * x * x);
  float ay = fabsf(y);
  float t  = __expf(-2.0f * ay);
  float th = 1.0f - __fdividef(2.0f * t, 1.0f + t);   // tanh(|y|)
  th = copysignf(th, y);
  return 0.5f * x * (1.0f + th);
}

struct Frag2 { uint4 a, b; };

// A fragment (16x32 bf16, M x K), source row-major [m][k], base pre-offset to (m0,k0).
static __device__ __forceinline__ v16bf load_fragA(const unsigned short* base, int pitch) {
  int lane = (int)(threadIdx.x & 31u);
  const unsigned short* p = base + (lane & 15) * pitch + (lane >> 4) * 8;
  Frag2 t;
  t.a = *(const uint4*)(p);
  t.b = *(const uint4*)(p + 16);
  return __builtin_bit_cast(v16bf, t);
}

// B fragment (32x16 bf16, K x N), source stored transposed [n][k], base pre-offset to (n0,k0).
static __device__ __forceinline__ v16bf load_fragB(const unsigned short* base, int pitch) {
  int lane = (int)(threadIdx.x & 31u);
  const unsigned short* p = base + (lane & 15) * pitch + (lane >> 4) * 8;
  Frag2 t;
  t.a = *(const uint4*)(p);
  t.b = *(const uint4*)(p + 16);
  return __builtin_bit_cast(v16bf, t);
}

static __device__ __forceinline__ v8f wmma_bf16(v16bf a, v16bf b, v8f c) {
  return __builtin_amdgcn_wmma_f32_16x16x32_bf16(false, a, false, b, (short)0, c, false, false);
}

// --- weight tile staging: async DMA to LDS when available, else via regs ----
struct W1Stage {              // 64 n-rows x 416 k  ->  LDS pitch 424
  uint4 v[7];
  __device__ __forceinline__ void load(const unsigned short* w1t, int s, int nh, int tid) {
    const unsigned short* src = w1t + (size_t)(s * 128 + nh * 64) * 416;
#pragma unroll
    for (int i = 0; i < 7; ++i) {
      int c = tid + i * 256;
      if (c < 64 * 26) {
        int n = c / 26, ko = c % 26;
        v[i] = *(const uint4*)(src + n * 416 + ko * 8);
      }
    }
  }
  __device__ __forceinline__ void store(unsigned short* buf, int tid) {
#pragma unroll
    for (int i = 0; i < 7; ++i) {
      int c = tid + i * 256;
      if (c < 64 * 26) {
        int n = c / 26, ko = c % 26;
        *(uint4*)(buf + n * 424 + ko * 8) = v[i];
      }
    }
  }
};
struct W2Stage {              // 80 n-rows x 128 k  ->  LDS pitch 136
  uint4 v[3];
  __device__ __forceinline__ void load(const unsigned short* w2t, int s, int tid) {
    const unsigned short* src = w2t + (size_t)s * 80 * 128;
#pragma unroll
    for (int i = 0; i < 3; ++i) {
      int c = tid + i * 256;
      if (c < 80 * 8) {
        int n = c / 8, ko = c % 8;
        v[i] = *(const uint4*)(src + n * 128 + ko * 8);
      }
    }
  }
  __device__ __forceinline__ void store(unsigned short* buf, int tid) {
#pragma unroll
    for (int i = 0; i < 3; ++i) {
      int c = tid + i * 256;
      if (c < 80 * 8) {
        int n = c / 8, ko = c % 8;
        *(uint4*)(buf + n * 136 + ko * 8) = v[i];
      }
    }
  }
};

#if USE_ASYNC_LDS
static __device__ __forceinline__ void stage_w1_async(const unsigned short* w1t, int s, int nh,
                                                      unsigned short* buf, int tid) {
  unsigned short* src = const_cast<unsigned short*>(w1t) + (size_t)(s * 128 + nh * 64) * 416;
#pragma unroll
  for (int i = 0; i < 7; ++i) {
    int c = tid + i * 256;
    if (c < 64 * 26) {
      int n = c / 26, ko = c % 26;
      __builtin_amdgcn_global_load_async_to_lds_b128(
          (as1_v4i)(v4i*)(src + n * 416 + ko * 8),
          (as3_v4i)(v4i*)(buf + n * 424 + ko * 8), 0, 0);
    }
  }
}
static __device__ __forceinline__ void stage_w2_async(const unsigned short* w2t, int s,
                                                      unsigned short* buf, int tid) {
  unsigned short* src = const_cast<unsigned short*>(w2t) + (size_t)s * 80 * 128;
#pragma unroll
  for (int i = 0; i < 3; ++i) {
    int c = tid + i * 256;
    if (c < 80 * 8) {
      int n = c / 8, ko = c % 8;
      __builtin_amdgcn_global_load_async_to_lds_b128(
          (as1_v4i)(v4i*)(src + n * 128 + ko * 8),
          (as3_v4i)(v4i*)(buf + n * 136 + ko * 8), 0, 0);
    }
  }
}
#endif

// ---------------------------------------------------------------------------
// Kernel 1: weight prep — bf16 transposed/padded copies + routing softmax.
// ---------------------------------------------------------------------------
__global__ void prep_kernel(const float* qe_w1, const float* qe_w2,
                            const float* keys, const float* stsub,
                            const float* hw1, const float* hw2,
                            unsigned short* w1t, unsigned short* w2t,
                            unsigned short* keysb, unsigned short* qw1t,
                            unsigned short* qw2t, float* smap) {
  const int N1 = 8 * 128 * 416;
  const int N2 = 8 * 80 * 128;
  const int N3 = 512 * 64;
  const int N4 = 64 * 32;
  const int N5 = 64 * 64;
  const int NC = N1 + N2 + N3 + N4 + N5;
  const int NT = NC + 512;
  int stride = (int)(gridDim.x * blockDim.x);
  for (int i = (int)(blockIdx.x * blockDim.x + threadIdx.x); i < NT; i += stride) {
    if (i < N1) {
      int k = i % 416, t = i / 416;
      int n = t % 128, s = t / 128;
      float v = (k < 400) ? hw1[(s * 400 + k) * 128 + n] : 0.f;
      w1t[i] = f2bf(v);
    } else if (i < N1 + N2) {
      int j = i - N1;
      int k = j % 128, t = j / 128;
      int n = t % 80, s = t / 80;
      float v = (n < 65) ? hw2[(s * 128 + k) * 65 + n] : 0.f;
      w2t[j] = f2bf(v);
    } else if (i < N1 + N2 + N3) {
      int j = i - (N1 + N2);
      keysb[j] = f2bf(keys[j]);
    } else if (i < N1 + N2 + N3 + N4) {
      int j = i - (N1 + N2 + N3);
      int k = j % 32, n = j / 32;
      qw1t[j] = f2bf((k < 16) ? qe_w1[k * 64 + n] : 0.f);
    } else if (i < NC) {
      int j = i - (N1 + N2 + N3 + N4);
      int k = j % 64, n = j / 64;
      qw2t[j] = f2bf(qe_w2[k * 64 + n]);
    } else {
      int r = i - NC;
      float v[8];
      float m = -3.0e38f;
      for (int c = 0; c < 8; ++c) { v[c] = stsub[r * 8 + c]; m = fmaxf(m, v[c]); }
      float ssum = 0.f;
      for (int c = 0; c < 8; ++c) { v[c] = __expf(v[c] - m); ssum += v[c]; }
      float inv = 1.f / ssum;
      for (int c = 0; c < 8; ++c) smap[r * 8 + c] = v[c] * inv;
    }
  }
}

// ---------------------------------------------------------------------------
// Kernel 2: encoder + router. 64 rows/block, 128 threads (4 waves).
// ---------------------------------------------------------------------------
__global__ void __launch_bounds__(128) router_kernel(
    const float* query, const float* qe_b1, const float* qe_b2,
    const float* temp,
    const unsigned short* qw1t, const unsigned short* qw2t,
    const unsigned short* keysb, const float* smap, float* subw) {
  __shared__ __align__(16) unsigned char smem[155136];
  float* sims = (float*)smem;                               // [64][514] f32
  unsigned short* aq  = (unsigned short*)(smem + 131584);   // [64][40] bf16
  unsigned short* h1b = (unsigned short*)(smem + 136704);   // [64][72] bf16
  unsigned short* qb  = (unsigned short*)(smem + 145920);   // [64][72] bf16
  // late-phase aliases (after GEMM3, aq/h1b/qb dead):
  float* smapl  = (float*)(smem + 131584);                  // [512][8]  -> ends 147968
  float* redm   = (float*)(smem + 147968);                  // [128]
  float* reds   = (float*)(smem + 148480);                  // [128]
  float* redacc = (float*)(smem + 148992);                  // [128][8]  -> ends 153088

  const int tid = (int)threadIdx.x;
  const int lane = tid & 31, wv = tid >> 5;
  const int hl = lane >> 4;
  const int row0 = (int)blockIdx.x * 64;
  const float invT = 1.0f / temp[0];

  for (int e = tid; e < 64 * 40; e += 128) {
    int r = e / 40, k = e % 40;
    aq[e] = f2bf((k < 16) ? query[(row0 + r) * 16 + k] : 0.f);
  }
  __syncthreads();

  const int m0 = wv * 16;

  // GEMM1: [64,32] x [32,64] -> GELU -> h1b
  {
    v16bf a = load_fragA(aq + m0 * 40, 40);
    for (int nt = 0; nt < 4; ++nt) {
      v8f d = {};
      d = wmma_bf16(a, load_fragB(qw1t + nt * 16 * 32, 32), d);
      int col = nt * 16 + (lane & 15);
      float b1 = qe_b1[col];
      for (int i = 0; i < 8; ++i) {
        int r = m0 + i + hl * 8;
        h1b[r * 72 + col] = f2bf(gelu_fast(d[i] + b1));
      }
    }
  }
  __syncthreads();

  // GEMM2: [64,64] x [64,64] -> q -> qb
  {
    v16bf a0 = load_fragA(h1b + m0 * 72 + 0, 72);
    v16bf a1 = load_fragA(h1b + m0 * 72 + 32, 72);
    for (int nt = 0; nt < 4; ++nt) {
      v8f d = {};
      d = wmma_bf16(a0, load_fragB(qw2t + nt * 16 * 64 + 0, 64), d);
      d = wmma_bf16(a1, load_fragB(qw2t + nt * 16 * 64 + 32, 64), d);
      int col = nt * 16 + (lane & 15);
      float b2 = qe_b2[col];
      for (int i = 0; i < 8; ++i) {
        int r = m0 + i + hl * 8;
        qb[r * 72 + col] = f2bf(d[i] + b2);
      }
    }
  }
  __syncthreads();

  // GEMM3: [64,64] x [64,512] -> sims (scaled by 1/temperature)
  {
    v16bf a0 = load_fragA(qb + m0 * 72 + 0, 72);
    v16bf a1 = load_fragA(qb + m0 * 72 + 32, 72);
    for (int n = 0; n < 32; ++n) {
      v8f d = {};
      d = wmma_bf16(a0, load_fragB(keysb + n * 16 * 64 + 0, 64), d);
      d = wmma_bf16(a1, load_fragB(keysb + n * 16 * 64 + 32, 64), d);
      int col = n * 16 + (lane & 15);
      for (int i = 0; i < 8; ++i) {
        int r = m0 + i + hl * 8;
        sims[r * 514 + col] = d[i] * invT;
      }
    }
  }
  __syncthreads();

  for (int e = tid; e < 512 * 8; e += 128) smapl[e] = smap[e];
  __syncthreads();

  // softmax(512) fused with x sub_map; 2 threads per row, LDS merge.
  {
    const int r = tid & 63, hh = tid >> 6;          // row, half-of-columns
    const float* sr = sims + r * 514 + hh * 256;
    float m = -3.0e38f;
    for (int j = 0; j < 256; ++j) m = fmaxf(m, sr[j]);
    redm[tid] = m;
    __syncthreads();
    m = fmaxf(redm[r], redm[r + 64]);
    float ssum = 0.f;
    float acc[8];
    for (int c = 0; c < 8; ++c) acc[c] = 0.f;
    const int j0 = hh * 256;
    for (int j = 0; j < 256; ++j) {
      float p = __expf(sr[j] - m);
      ssum += p;
      const float* mp = smapl + (j0 + j) * 8;
      for (int c = 0; c < 8; ++c) acc[c] += p * mp[c];
    }
    reds[tid] = ssum;
    for (int c = 0; c < 8; ++c) redacc[tid * 8 + c] = acc[c];
    __syncthreads();
    if (tid < 64) {
      float inv = 1.f / (reds[tid] + reds[tid + 64]);
      float* o = subw + (size_t)(row0 + tid) * 8;
      for (int c = 0; c < 8; ++c)
        o[c] = (redacc[tid * 8 + c] + redacc[(tid + 64) * 8 + c]) * inv;
    }
  }
}

// ---------------------------------------------------------------------------
// Kernel 3: handler MLPs. 128 rows/block, 256 threads (8 waves).
// Double-buffered weight tiles; staging is async-DMA to the alternate LDS
// buffer (overlaps the WMMA chain), with s_wait_asynccnt 0 before the barrier.
// ---------------------------------------------------------------------------
__global__ void __launch_bounds__(256) handler_kernel(
    const float* query, const float* args,
    const float* hb1, const float* hb2,
    const unsigned short* w1t, const unsigned short* w2t,
    const float* subw, float* out) {
  __shared__ __align__(16) unsigned char smem[256000];
  unsigned short* hinb  = (unsigned short*)(smem);           // [128][424] bf16
  unsigned short* buf0  = (unsigned short*)(smem + 108544);  // w1:[64][424] / w2:[80][136]
  unsigned short* buf1  = (unsigned short*)(smem + 162816);  // (second buffer)
  unsigned short* H     = (unsigned short*)(smem + 217088);  // [128][136] bf16
  float* subwl          = (float*)(smem + 251904);           // [128][8] f32
  unsigned short* bufs[2] = { buf0, buf1 };

  const int tid = (int)threadIdx.x;
  const int lane = tid & 31, wv = tid >> 5;
  const int hl = lane >> 4;
  const int row0 = (int)blockIdx.x * 128;
  const int m0 = wv * 16;

  // initial w1(s=0, half=0) stage into buf0 (async: overlaps the hin fill)
#if USE_ASYNC_LDS
  stage_w1_async(w1t, 0, 0, buf0, tid);
#endif
  // stage hin = concat(query, args), bf16, K padded 400 -> 416 (..423 zero)
  for (int e = tid; e < 128 * 424; e += 256) {
    int r = e / 424, k = e % 424;
    float v;
    if (k < 16)       v = query[(row0 + r) * 16 + k];
    else if (k < 400) v = args[(row0 + r) * 384 + (k - 16)];
    else              v = 0.f;
    hinb[e] = f2bf(v);
  }
  for (int e = tid; e < 128 * 8; e += 256) subwl[e] = subw[(size_t)row0 * 8 + e];
#if USE_ASYNC_LDS
  __builtin_amdgcn_s_wait_asynccnt(0);
#else
  {
    W1Stage st;
    st.load(w1t, 0, 0, tid);
    st.store(buf0, tid);
  }
#endif
  __syncthreads();

  v8f acc[5] = {};   // 16 rows x 80 cols per wave, f32
  int cur = 0;

  for (int s = 0; s < 8; ++s) {
    // ---- phase 1: GEMM1 half 0 from bufs[cur]; prefetch w1(s,1) -> bufs[cur^1]
    {
      W1Stage st;
#if USE_ASYNC_LDS
      stage_w1_async(w1t, s, 1, bufs[cur ^ 1], tid);
#else
      st.load(w1t, s, 1, tid);                    // VMEM issued before WMMA chain
#endif
      const unsigned short* wt = bufs[cur];
      v8f d0 = {}, d1 = {}, d2 = {}, d3 = {};
      for (int k0 = 0; k0 < 416; k0 += 32) {
        v16bf a = load_fragA(hinb + m0 * 424 + k0, 424);
        d0 = wmma_bf16(a, load_fragB(wt + 0 * 16 * 424 + k0, 424), d0);
        d1 = wmma_bf16(a, load_fragB(wt + 1 * 16 * 424 + k0, 424), d1);
        d2 = wmma_bf16(a, load_fragB(wt + 2 * 16 * 424 + k0, 424), d2);
        d3 = wmma_bf16(a, load_fragB(wt + 3 * 16 * 424 + k0, 424), d3);
      }
      v8f dd[4] = {d0, d1, d2, d3};
      for (int nt = 0; nt < 4; ++nt) {
        int col = nt * 16 + (lane & 15);
        float b1 = hb1[s * 128 + col];
        for (int i = 0; i < 8; ++i) {
          int r = m0 + i + hl * 8;
          H[r * 136 + col] = f2bf(gelu_fast(dd[nt][i] + b1));
        }
      }
#if USE_ASYNC_LDS
      __builtin_amdgcn_s_wait_asynccnt(0);
#else
      st.store(bufs[cur ^ 1], tid);
#endif
    }
    __syncthreads();
    cur ^= 1;
    // ---- phase 2: GEMM1 half 1 from bufs[cur]; prefetch w2(s) -> bufs[cur^1]
    {
      W2Stage st;
#if USE_ASYNC_LDS
      stage_w2_async(w2t, s, bufs[cur ^ 1], tid);
#else
      st.load(w2t, s, tid);
#endif
      const unsigned short* wt = bufs[cur];
      v8f d0 = {}, d1 = {}, d2 = {}, d3 = {};
      for (int k0 = 0; k0 < 416; k0 += 32) {
        v16bf a = load_fragA(hinb + m0 * 424 + k0, 424);
        d0 = wmma_bf16(a, load_fragB(wt + 0 * 16 * 424 + k0, 424), d0);
        d1 = wmma_bf16(a, load_fragB(wt + 1 * 16 * 424 + k0, 424), d1);
        d2 = wmma_bf16(a, load_fragB(wt + 2 * 16 * 424 + k0, 424), d2);
        d3 = wmma_bf16(a, load_fragB(wt + 3 * 16 * 424 + k0, 424), d3);
      }
      v8f dd[4] = {d0, d1, d2, d3};
      for (int nt = 0; nt < 4; ++nt) {
        int col = 64 + nt * 16 + (lane & 15);
        float b1 = hb1[s * 128 + col];
        for (int i = 0; i < 8; ++i) {
          int r = m0 + i + hl * 8;
          H[r * 136 + col] = f2bf(gelu_fast(dd[nt][i] + b1));
        }
      }
#if USE_ASYNC_LDS
      __builtin_amdgcn_s_wait_asynccnt(0);
#else
      st.store(bufs[cur ^ 1], tid);
#endif
    }
    __syncthreads();
    cur ^= 1;
    // ---- phase 3: GEMM2 from bufs[cur] (w2 tile); prefetch w1(s+1,0) -> bufs[cur^1]
    {
      W1Stage st;
#if USE_ASYNC_LDS
      if (s < 7) stage_w1_async(w1t, s + 1, 0, bufs[cur ^ 1], tid);
#else
      if (s < 7) st.load(w1t, s + 1, 0, tid);
#endif
      const unsigned short* wt = bufs[cur];
      v16bf a0 = load_fragA(H + m0 * 136 + 0,  136);
      v16bf a1 = load_fragA(H + m0 * 136 + 32, 136);
      v16bf a2 = load_fragA(H + m0 * 136 + 64, 136);
      v16bf a3 = load_fragA(H + m0 * 136 + 96, 136);
      for (int nt = 0; nt < 5; ++nt) {
        v8f d = {};
        d = wmma_bf16(a0, load_fragB(wt + nt * 16 * 136 + 0,  136), d);
        d = wmma_bf16(a1, load_fragB(wt + nt * 16 * 136 + 32, 136), d);
        d = wmma_bf16(a2, load_fragB(wt + nt * 16 * 136 + 64, 136), d);
        d = wmma_bf16(a3, load_fragB(wt + nt * 16 * 136 + 96, 136), d);
        int col = nt * 16 + (lane & 15);
        float b2 = (col < 65) ? hb2[s * 65 + col] : 0.f;
        for (int i = 0; i < 8; ++i) {
          int r = m0 + i + hl * 8;
          float sw = subwl[r * 8 + s];
          acc[nt][i] += sw * (d[i] + b2);
        }
      }
#if USE_ASYNC_LDS
      __builtin_amdgcn_s_wait_asynccnt(0);
#else
      if (s < 7) st.store(bufs[cur ^ 1], tid);
#endif
    }
    __syncthreads();
    cur ^= 1;
  }

  // final store (cols 0..64)
  for (int nt = 0; nt < 5; ++nt) {
    int col = nt * 16 + (lane & 15);
    if (col < 65) {
      for (int i = 0; i < 8; ++i) {
        int r = row0 + m0 + i + hl * 8;
        out[(size_t)r * 65 + col] = acc[nt][i];
      }
    }
  }
}

// ---------------------------------------------------------------------------
// Launch
// ---------------------------------------------------------------------------
extern "C" void kernel_launch(void* const* d_in, const int* in_sizes, int n_in,
                              void* d_out, int out_size, void* d_ws, size_t ws_size,
                              hipStream_t stream) {
  (void)in_sizes; (void)n_in; (void)out_size; (void)ws_size;
  const float* query = (const float*)d_in[0];
  const float* args  = (const float*)d_in[1];
  const float* qe_w1 = (const float*)d_in[2];
  const float* qe_b1 = (const float*)d_in[3];
  const float* qe_w2 = (const float*)d_in[4];
  const float* qe_b2 = (const float*)d_in[5];
  const float* keys  = (const float*)d_in[6];
  const float* stsub = (const float*)d_in[7];
  const float* temp  = (const float*)d_in[8];
  const float* hw1   = (const float*)d_in[9];
  const float* hb1   = (const float*)d_in[10];
  const float* hw2   = (const float*)d_in[11];
  const float* hb2   = (const float*)d_in[12];
  float* out = (float*)d_out;

  unsigned char* ws = (unsigned char*)d_ws;
  unsigned short* w1t   = (unsigned short*)(ws + 0);        //   851968 B
  unsigned short* w2t   = (unsigned short*)(ws + 851968);   //   163840 B
  unsigned short* keysb = (unsigned short*)(ws + 1015808);  //    65536 B
  unsigned short* qw1t  = (unsigned short*)(ws + 1081344);  //     4096 B
  unsigned short* qw2t  = (unsigned short*)(ws + 1085440);  //     8192 B
  float*          smap  = (float*)(ws + 1093632);           //    16384 B
  float*          subw  = (float*)(ws + 1110016);           //  4194304 B  (total ~5.06 MB)

  prep_kernel<<<2139, 256, 0, stream>>>(qe_w1, qe_w2, keys, stsub, hw1, hw2,
                                        w1t, w2t, keysb, qw1t, qw2t, smap);
  router_kernel<<<BTOT / 64, 128, 0, stream>>>(query, qe_b1, qe_b2, temp,
                                               qw1t, qw2t, keysb, smap, subw);
  handler_kernel<<<BTOT / 128, 256, 0, stream>>>(query, args, hb1, hb2,
                                                 w1t, w2t, subw, out);
}
